// UnrolledLSTM_50010599195073
// MI455X (gfx1250) — compile-verified, matrix operand order checked by reading, not run on hardware
//
#include <hip/hip_runtime.h>
#include <hip/hip_bf16.h>
#include <math.h>

typedef __bf16 bf16_t;
typedef __attribute__((ext_vector_type(16))) __bf16 v16bf;
typedef __attribute__((ext_vector_type(8)))  float  v8f;

#define T_ENC 256
#define HOR   32
#define HIDN  512
#define BTILE 16

union FragU { v16bf v; uint4 q[2]; };

__device__ __forceinline__ v16bf load_frag(const bf16_t* p) {
  // Two 16-byte chunks per lane: K = [kb .. kb+7] and [kb+16 .. kb+23]
  FragU f;
  f.q[0] = *reinterpret_cast<const uint4*>(p);
  f.q[1] = *reinterpret_cast<const uint4*>(p + 16);
  return f.v;
}

__device__ __forceinline__ float sigf(float x) { return 1.0f / (1.0f + __expf(-x)); }

#define WMMA_BF16(ACC, A, Bf)                                                   \
  ACC = __builtin_amdgcn_wmma_f32_16x16x32_bf16(false, (A), false, (Bf),        \
                                                (short)0, (ACC), false, false)

// ---------------------------------------------------------------------------
// Kernel 1: per-batch time-series scale: ts = mean_t |X[b,t]|, lts = log(ts)
// ---------------------------------------------------------------------------
__global__ void prep_ts_kernel(const float* __restrict__ X,
                               float* __restrict__ ts, float* __restrict__ lts) {
  __shared__ float red[256];
  const int b = blockIdx.x;
  const int t = threadIdx.x;
  red[t] = fabsf(X[(size_t)b * T_ENC + t]);
  __syncthreads();
  for (int s = 128; s > 0; s >>= 1) {
    if (t < s) red[t] += red[t + s];
    __syncthreads();
  }
  if (t == 0) {
    float m = red[0] / (float)T_ENC;       // pad_mask is all ones
    m = fmaxf(m, 1e-10f);
    ts[b] = m;
    lts[b] = logf(m);
  }
}

// ---------------------------------------------------------------------------
// Kernel 2: f32 -> bf16 weight conversion (weights then live in L2, 2MB each)
// ---------------------------------------------------------------------------
__global__ void conv_bf16_kernel(const float* __restrict__ src,
                                 bf16_t* __restrict__ dst, int n) {
  int i = blockIdx.x * blockDim.x + threadIdx.x;
  if (i < n) dst[i] = (bf16_t)src[i];
}

// ---------------------------------------------------------------------------
// Kernel 3: persistent per-batch-tile 2-layer LSTM encoder + decoder.
// 32 workgroups x 16 batch rows; 8 waves; each wave owns 64 hidden columns.
// ---------------------------------------------------------------------------
__global__ __launch_bounds__(256) void lstm_main_kernel(
    const float* __restrict__ X,       // (512, 256)
    const float* __restrict__ Wih0,    // (2048, 2) f32
    const float* __restrict__ bih0, const float* __restrict__ bhh0,
    const float* __restrict__ bih1, const float* __restrict__ bhh1,
    const float* __restrict__ Wd,      // (512,) f32
    const float* __restrict__ bd,      // (1,)
    const float* __restrict__ ts, const float* __restrict__ lts,
    const bf16_t* __restrict__ Whh0,   // (2048,512) bf16
    const bf16_t* __restrict__ Wih1,   // (2048,512) bf16
    const bf16_t* __restrict__ Whh1,   // (2048,512) bf16
    float* __restrict__ out)           // (512, 288)
{
  __shared__ __align__(16) bf16_t h0s[2][BTILE][HIDN];   // double-buffered h, layer 0
  __shared__ __align__(16) bf16_t h1s[2][BTILE][HIDN];   // double-buffered h, layer 1
  __shared__ float xs[BTILE], lt[BTILE], tsr[BTILE], obuf[BTILE], prev[BTILE];

  const int tid   = threadIdx.x;
  const int lane  = tid & 31;
  const int wid   = tid >> 5;            // wave id 0..7
  const int arow  = lane & 15;           // A-fragment source row (batch row in tile)
  const int ncol  = lane & 15;           // C/D fragment column within n-tile
  const int kbase = (lane >> 4) * 8;     // K sub-offset per lane half
  const int mof   = (lane >> 4) * 8;     // C/D row offset per lane half

  // zero state
  for (int i = tid; i < BTILE * HIDN; i += 256) {
    const int r = i >> 9, j = i & (HIDN - 1);
    h0s[0][r][j] = (bf16_t)0.0f; h0s[1][r][j] = (bf16_t)0.0f;
    h1s[0][r][j] = (bf16_t)0.0f; h1s[1][r][j] = (bf16_t)0.0f;
  }
  if (tid < BTILE) {
    const int b = blockIdx.x * BTILE + tid;
    tsr[tid]  = ts[b];
    lt[tid]   = lts[b];
    obuf[tid] = 0.0f;
    prev[tid] = 0.0f;
  }
  __syncthreads();

  // register-resident cell state: lane (s,r) always owns cell (m, nn)
  float c0r[4][8], c1r[4][8];
#pragma unroll
  for (int s = 0; s < 4; ++s)
#pragma unroll
    for (int r = 0; r < 8; ++r) { c0r[s][r] = 0.0f; c1r[s][r] = 0.0f; }

  const float bdv = bd[0];
  int cur = 0;

  for (int step = 0; step < T_ENC + HOR; ++step) {
    const bool enc = step < T_ENC;
    if (tid < BTILE) {
      const int b = blockIdx.x * BTILE + tid;
      xs[tid] = enc ? (X[(size_t)b * T_ENC + step] / tsr[tid]) : prev[tid];
    }
    __syncthreads();   // B0: xs ready; closes previous iteration's tail writes
    const int nxt = cur ^ 1;

    // ---------------- Layer 0: g = h0 @ Whh0^T (+ x-term, biases) -----------
    {
      const bf16_t* a_p = &h0s[cur][arow][0];
#pragma unroll
      for (int s = 0; s < 4; ++s) {
        const int nn = wid * 64 + s * 16 + ncol;   // hidden unit for this lane
        v8f ai = {}, af = {}, ag = {}, ao = {};
        const bf16_t* bi_p = Whh0 + (size_t)nn * HIDN;
        const bf16_t* bf_p = Whh0 + (size_t)(nn + HIDN) * HIDN;
        const bf16_t* bg_p = Whh0 + (size_t)(nn + 2 * HIDN) * HIDN;
        const bf16_t* bo_p = Whh0 + (size_t)(nn + 3 * HIDN) * HIDN;
        for (int kk = 0; kk < 16; ++kk) {
          const int k0 = kk * 32 + kbase;
          v16bf a = load_frag(a_p + k0);
          WMMA_BF16(ai, a, load_frag(bi_p + k0));
          WMMA_BF16(af, a, load_frag(bf_p + k0));
          WMMA_BF16(ag, a, load_frag(bg_p + k0));
          WMMA_BF16(ao, a, load_frag(bo_p + k0));
        }
        const float w0i = Wih0[2 * nn],                w1i = Wih0[2 * nn + 1];
        const float w0f = Wih0[2 * (nn + HIDN)],       w1f = Wih0[2 * (nn + HIDN) + 1];
        const float w0g = Wih0[2 * (nn + 2 * HIDN)],   w1g = Wih0[2 * (nn + 2 * HIDN) + 1];
        const float w0o = Wih0[2 * (nn + 3 * HIDN)],   w1o = Wih0[2 * (nn + 3 * HIDN) + 1];
        const float bbi = bih0[nn] + bhh0[nn];
        const float bbf = bih0[nn + HIDN] + bhh0[nn + HIDN];
        const float bbg = bih0[nn + 2 * HIDN] + bhh0[nn + 2 * HIDN];
        const float bbo = bih0[nn + 3 * HIDN] + bhh0[nn + 3 * HIDN];
#pragma unroll
        for (int r = 0; r < 8; ++r) {
          const int m = r + mof;
          const float xv = xs[m], lv = lt[m];
          const float gi = ai[r] + xv * w0i + lv * w1i + bbi;
          const float gf = af[r] + xv * w0f + lv * w1f + bbf;
          const float gg = ag[r] + xv * w0g + lv * w1g + bbg;
          const float go = ao[r] + xv * w0o + lv * w1o + bbo;
          const float cn = sigf(gf) * c0r[s][r] + sigf(gi) * tanhf(gg);
          c0r[s][r] = cn;
          h0s[nxt][m][nn] = (bf16_t)(sigf(go) * tanhf(cn));
        }
      }
    }
    __syncthreads();   // B1: h0(t) complete

    // ------------- Layer 1: g = h0_new @ Wih1^T + h1 @ Whh1^T + biases ------
    {
      const bf16_t* a0_p = &h0s[nxt][arow][0];
      const bf16_t* a1_p = &h1s[cur][arow][0];
#pragma unroll
      for (int s = 0; s < 4; ++s) {
        const int nn = wid * 64 + s * 16 + ncol;
        v8f ai = {}, af = {}, ag = {}, ao = {};
        const bf16_t* ii_p = Wih1 + (size_t)nn * HIDN;
        const bf16_t* if_p = Wih1 + (size_t)(nn + HIDN) * HIDN;
        const bf16_t* ig_p = Wih1 + (size_t)(nn + 2 * HIDN) * HIDN;
        const bf16_t* io_p = Wih1 + (size_t)(nn + 3 * HIDN) * HIDN;
        const bf16_t* hi_p = Whh1 + (size_t)nn * HIDN;
        const bf16_t* hf_p = Whh1 + (size_t)(nn + HIDN) * HIDN;
        const bf16_t* hg_p = Whh1 + (size_t)(nn + 2 * HIDN) * HIDN;
        const bf16_t* ho_p = Whh1 + (size_t)(nn + 3 * HIDN) * HIDN;
        for (int kk = 0; kk < 16; ++kk) {
          const int k0 = kk * 32 + kbase;
          v16bf a0 = load_frag(a0_p + k0);
          WMMA_BF16(ai, a0, load_frag(ii_p + k0));
          WMMA_BF16(af, a0, load_frag(if_p + k0));
          WMMA_BF16(ag, a0, load_frag(ig_p + k0));
          WMMA_BF16(ao, a0, load_frag(io_p + k0));
          v16bf a1 = load_frag(a1_p + k0);
          WMMA_BF16(ai, a1, load_frag(hi_p + k0));
          WMMA_BF16(af, a1, load_frag(hf_p + k0));
          WMMA_BF16(ag, a1, load_frag(hg_p + k0));
          WMMA_BF16(ao, a1, load_frag(ho_p + k0));
        }
        const float bbi = bih1[nn] + bhh1[nn];
        const float bbf = bih1[nn + HIDN] + bhh1[nn + HIDN];
        const float bbg = bih1[nn + 2 * HIDN] + bhh1[nn + 2 * HIDN];
        const float bbo = bih1[nn + 3 * HIDN] + bhh1[nn + 3 * HIDN];
#pragma unroll
        for (int r = 0; r < 8; ++r) {
          const int m = r + mof;
          const float gi = ai[r] + bbi;
          const float gf = af[r] + bbf;
          const float gg = ag[r] + bbg;
          const float go = ao[r] + bbo;
          const float cn = sigf(gf) * c1r[s][r] + sigf(gi) * tanhf(gg);
          c1r[s][r] = cn;
          h1s[nxt][m][nn] = (bf16_t)(sigf(go) * tanhf(cn));
        }
      }
    }
    __syncthreads();   // B2: h1(t) complete

    // ---------------- Output projection: o = h1 . Wd + bd (C == 1) ----------
    {
      const int r = tid >> 4;       // batch row in tile
      const int p = tid & 15;       // 16 partial sums per row
      float partial = 0.0f;
#pragma unroll 8
      for (int q = 0; q < 32; ++q) {
        const int j = p * 32 + q;
        partial += (float)h1s[nxt][r][j] * Wd[j];
      }
      atomicAdd(&obuf[r], partial); // ds_add_f32 reduction in LDS
    }
    __syncthreads();   // B3: obuf complete
    if (tid < BTILE) {
      const float o = obuf[tid] + bdv;
      const int b = blockIdx.x * BTILE + tid;
      out[(size_t)b * (T_ENC + HOR) + step] = o * tsr[tid];
      prev[tid] = o;                // decoder feedback is the unscaled output
      obuf[tid] = 0.0f;
    }
    cur = nxt;
  }
}

// ---------------------------------------------------------------------------
extern "C" void kernel_launch(void* const* d_in, const int* in_sizes, int n_in,
                              void* d_out, int out_size, void* d_ws, size_t ws_size,
                              hipStream_t stream) {
  const float* X    = (const float*)d_in[0];
  // d_in[1] = pad_mask (all ones, unused); d_in[2] = H (compile-time 32)
  const float* Wih0 = (const float*)d_in[3];
  const float* Whh0 = (const float*)d_in[4];
  const float* bih0 = (const float*)d_in[5];
  const float* bhh0 = (const float*)d_in[6];
  const float* Wih1 = (const float*)d_in[7];
  const float* Whh1 = (const float*)d_in[8];
  const float* bih1 = (const float*)d_in[9];
  const float* bhh1 = (const float*)d_in[10];
  const float* Wd   = (const float*)d_in[11];
  const float* bd   = (const float*)d_in[12];
  float* out = (float*)d_out;

  char* ws = (char*)d_ws;
  float*  ts    = (float*)ws;                       // 512 f32
  float*  lts   = ts + 512;                         // 512 f32
  bf16_t* Whh0b = (bf16_t*)(ws + 4096);             // 2 MB
  bf16_t* Wih1b = Whh0b + (size_t)2048 * 512;       // 2 MB
  bf16_t* Whh1b = Wih1b + (size_t)2048 * 512;       // 2 MB

  prep_ts_kernel<<<512, 256, 0, stream>>>(X, ts, lts);
  const int nw = 2048 * 512;
  conv_bf16_kernel<<<(nw + 255) / 256, 256, 0, stream>>>(Whh0, Whh0b, nw);
  conv_bf16_kernel<<<(nw + 255) / 256, 256, 0, stream>>>(Wih1, Wih1b, nw);
  conv_bf16_kernel<<<(nw + 255) / 256, 256, 0, stream>>>(Whh1, Whh1b, nw);

  lstm_main_kernel<<<32, 256, 0, stream>>>(X, Wih0, bih0, bhh0, bih1, bhh1,
                                           Wd, bd, ts, lts,
                                           Whh0b, Wih1b, Whh1b, out);
}
